// CalibratedGateModel_56109452755473
// MI455X (gfx1250) — compile-verified, matrix operand order checked by reading, not run on hardware
//
#include <hip/hip_runtime.h>

// Problem constants
#define Bq  32
#define Lq  2048
#define Hq  128
#define H2q 256
#define Vq  32000

typedef __attribute__((ext_vector_type(16))) _Float16 v16h;
typedef __attribute__((ext_vector_type(8)))  _Float16 v8h;
typedef __attribute__((ext_vector_type(8)))  float    v8f;
typedef __attribute__((ext_vector_type(4)))  unsigned int u32x4;
typedef __attribute__((ext_vector_type(8)))  int      i32x8;
typedef __attribute__((ext_vector_type(4)))  int      i32x4;

#if __has_builtin(__builtin_amdgcn_tensor_load_to_lds) && \
    __has_builtin(__builtin_amdgcn_s_wait_tensorcnt)
#define HAVE_TDM 1
#endif

// LDS layout for fused_token_kernel (dynamic), bytes:
#define OFF_LW1 0         //  64 KB : ff_w1 f16 fragments (256x128)
#define OFF_LW2 65536     //  64 KB : ff_w2 f16 fragments (128x256)
#define OFF_LKP 131072    //  32 KB : kp_w  f16 fragments (128x128)
#define OFF_XS  163840    //  64 KB : 8 waves x 16x128 f32
#define OFF_F1  229376    //  64 KB : 8 waves x 256x16 f16 (column-major)
#define SMEM_A  294912

// ---------------------------------------------------------------------------
// WMMA helpers (16x16x32 f16 -> f32).  C/D layout per CDNA5 ISA:
//   lane<16 : N=lane, M=vgpr r ; lane>=16 : N=lane-16, M=vgpr r + 8
// ---------------------------------------------------------------------------
__device__ __forceinline__ v8f wmma16(v16h a, v16h b, v8f c) {
  return __builtin_amdgcn_wmma_f32_16x16x32_f16(false, a, false, b,
                                                (short)0, c, false, false);
}

__device__ __forceinline__ v16h frag_a_f32(const float* base, int ld,
                                           int kbase, int lane) {
  const int row  = lane & 15;
  const int koff = (lane < 16) ? 0 : 8;
  const float* p = base + row * ld + kbase + koff;
  v16h a;
#pragma unroll
  for (int i = 0; i < 8; ++i) a[i] = (_Float16)p[i];
#pragma unroll
  for (int i = 0; i < 8; ++i) a[8 + i] = (_Float16)p[16 + i];
  return a;
}

// A fragment from column-major f16 buffer: element (row,k) at k*16 + row.
__device__ __forceinline__ v16h frag_a_cm16(const _Float16* base, int kbase,
                                            int lane) {
  const int row  = lane & 15;
  const int koff = (lane < 16) ? 0 : 8;
  const _Float16* p = base + (kbase + koff) * 16 + row;
  v16h a;
#pragma unroll
  for (int i = 0; i < 8; ++i) a[i] = p[i * 16];
#pragma unroll
  for (int i = 0; i < 8; ++i) a[8 + i] = p[(16 + i) * 16];
  return a;
}

// Pre-swizzled B fragment: lane's 16 halves are contiguous (32B -> 2x b128).
__device__ __forceinline__ v16h frag_ld(const _Float16* buf, int tile, int lane) {
  return *(const v16h*)(buf + (((size_t)tile * 32 + lane) << 4));
}

// weights stored (N, K) row-major in f32; fragment of B = W^T (K x 16)
__device__ __forceinline__ v16h frag_bT_f32(const float* w, int ldk,
                                            int nbase, int kbase, int lane) {
  const int col  = lane & 15;
  const int koff = (lane < 16) ? 0 : 8;
  const float* p = w + (size_t)(nbase + col) * ldk + kbase + koff;
  v16h b;
#pragma unroll
  for (int i = 0; i < 8; ++i) b[i] = (_Float16)p[i];
#pragma unroll
  for (int i = 0; i < 8; ++i) b[8 + i] = (_Float16)p[16 + i];
  return b;
}

#if defined(HAVE_TDM)
// 1D TDM copy: ndw dwords global -> LDS.  D# per CDNA5 ISA.
__device__ __forceinline__ void tdm_copy_1d(unsigned lds_addr, const void* gsrc,
                                            unsigned ndw) {
  const unsigned long long ga = (unsigned long long)(size_t)gsrc;
  u32x4 g0;
  g0[0] = 1u;                                     // count = 1 valid descriptor
  g0[1] = lds_addr;                               // LDS byte address
  g0[2] = (unsigned)(ga & 0xFFFFFFFFu);           // global addr lo
  g0[3] = (unsigned)((ga >> 32) & 0x1FFFFFFu) | (2u << 30);  // hi | type=image
  i32x8 g1;
  g1[0] = (int)(2u << 16);                        // data_size = 4B
  g1[1] = (int)((ndw & 0xFFFFu) << 16);           // tensor_dim0 lo16
  g1[2] = (int)(((ndw >> 16) & 0xFFFFu) | (1u << 16)); // dim0 hi | tensor_dim1=1
  g1[3] = (int)((ndw & 0xFFFFu) << 16);           // tile_dim0 (= ndw <= 16384)
  g1[4] = 1;                                      // tile_dim1 = 1
  g1[5] = (int)ndw;                               // tensor_dim0_stride lo
  g1[6] = 0;
  g1[7] = 0;
  i32x4 z4 = {};
#if __clang_major__ >= 23
  i32x8 z8 = {};
  __builtin_amdgcn_tensor_load_to_lds(g0, g1, z4, z4, z8, 0);
#else
  __builtin_amdgcn_tensor_load_to_lds(g0, g1, z4, z4, 0);
#endif
}
#endif

// ---------------------------------------------------------------------------
// Prep kernel: f32 (N,K) row-major weights -> f16 WMMA-B fragment buffers.
// Fragment layout: tile = nt*(K/32)+kc ; within tile: [lane][16 halves].
// ---------------------------------------------------------------------------
__global__ void swizzle_w_kernel(const float* __restrict__ w, int N, int K,
                                 _Float16* __restrict__ out) {
  const int idx = blockIdx.x * blockDim.x + threadIdx.x;
  if (idx >= N * K) return;
  const int tile = idx >> 9;          // / 512 halves per tile
  const int lane = (idx >> 4) & 31;
  const int i    = idx & 15;
  const int kch  = K >> 5;
  const int nt   = tile / kch;
  const int kc   = tile - nt * kch;
  const int col  = lane & 15;
  const int koff = (lane < 16) ? 0 : 8;
  const int k    = kc * 32 + ((i < 8) ? (koff + i) : (16 + koff + (i - 8)));
  const int n    = nt * 16 + col;
  out[idx] = (_Float16)w[(size_t)n * K + k];
}

// ---------------------------------------------------------------------------
// Kernel A: embed gather -> FF1(relu) -> FF2 + residual -> LN -> kp (WMMA).
// 8 waves/block, each wave owns a 16-token tile.  Pre-swizzled f16 weights are
// staged global->LDS once per block via the Tensor Data Mover, overlapped with
// the embedding gather; every B fragment afterwards is a 32B LDS vector load.
// f1 (relu output) is kept column-major so each lane's 8 accumulator values
// store as a single packed ds_store_b128.
// ---------------------------------------------------------------------------
__global__ void fused_token_kernel(const int* __restrict__ seq,
                                   const float* __restrict__ embed_w,
                                   const _Float16* __restrict__ w1f,
                                   const float* __restrict__ ff_b1,
                                   const _Float16* __restrict__ w2f,
                                   const float* __restrict__ ff_b2,
                                   const float* __restrict__ ln_g,
                                   const float* __restrict__ ln_b,
                                   const _Float16* __restrict__ kpf,
                                   float* __restrict__ k_all) {
  extern __shared__ char smem[];
  _Float16* lw1 = (_Float16*)(smem + OFF_LW1);
  _Float16* lw2 = (_Float16*)(smem + OFF_LW2);
  _Float16* lkp = (_Float16*)(smem + OFF_LKP);
  const int wave = threadIdx.x >> 5;
  const int lane = threadIdx.x & 31;
  float*    xs = (float*)(smem + OFF_XS) + wave * (16 * Hq);
  _Float16* f1 = (_Float16*)(smem + OFF_F1) + wave * (16 * H2q); // [256][16] cm

  // --- stage weights into LDS (TDM async; fallback: cooperative copy) ------
#if defined(HAVE_TDM)
  if (threadIdx.x < 32) {
    tdm_copy_1d((unsigned)(size_t)(const void*)lw1, w1f, (256 * 128) / 2);
    tdm_copy_1d((unsigned)(size_t)(const void*)lw2, w2f, (128 * 256) / 2);
    tdm_copy_1d((unsigned)(size_t)(const void*)lkp, kpf, (128 * 128) / 2);
  }
#else
  {
    const unsigned* s1 = (const unsigned*)w1f; unsigned* d1 = (unsigned*)lw1;
    for (int i = threadIdx.x; i < 16384; i += 256) d1[i] = s1[i];
    const unsigned* s2 = (const unsigned*)w2f; unsigned* d2 = (unsigned*)lw2;
    for (int i = threadIdx.x; i < 16384; i += 256) d2[i] = s2[i];
    const unsigned* s3 = (const unsigned*)kpf; unsigned* d3 = (unsigned*)lkp;
    for (int i = threadIdx.x; i < 8192;  i += 256) d3[i] = s3[i];
  }
#endif

  const int tile = blockIdx.x * 8 + wave;
  const int tok0 = tile * 16;

  // --- embedding gather into xs (f32), overlapped with TDM ----------------
  for (int i = lane; i < 16 * 32; i += 32) {
    const int r = i >> 5, c4 = i & 31;
    const int tok = seq[tok0 + r];
    ((float4*)xs)[r * 32 + c4] = ((const float4*)embed_w)[(size_t)tok * 32 + c4];
  }
#if defined(HAVE_TDM)
  __builtin_amdgcn_s_wait_tensorcnt(0);
#endif
  __syncthreads();

  v16h ah[4];
#pragma unroll
  for (int kc = 0; kc < 4; ++kc) ah[kc] = frag_a_f32(xs, Hq, kc * 32, lane);

  const int ncol  = lane & 15;
  const int mbase = (lane < 16) ? 0 : 8;

  // --- FF1 + ReLU -> f1 (f16, column-major, one b128 store per n-tile) ----
  for (int nt = 0; nt < 16; ++nt) {
    const int n = nt * 16 + ncol;
    const float b1 = ff_b1[n];
    v8f c;
#pragma unroll
    for (int r = 0; r < 8; ++r) c[r] = b1;
#pragma unroll
    for (int kc = 0; kc < 4; ++kc)
      c = wmma16(ah[kc], frag_ld(lw1, nt * 4 + kc, lane), c);
    v8h pk;
#pragma unroll
    for (int r = 0; r < 8; ++r) pk[r] = (_Float16)fmaxf(c[r], 0.0f);
    *(v8h*)(f1 + n * 16 + mbase) = pk;     // rows mbase..mbase+7 of column n
  }
  __syncthreads();

  // --- FF2 + residual -> xs  (A fragments hoisted: invariant of n-tile) ----
  v16h af[8];
#pragma unroll
  for (int kc = 0; kc < 8; ++kc) af[kc] = frag_a_cm16(f1, kc * 32, lane);
  for (int nt = 0; nt < 8; ++nt) {
    const int n = nt * 16 + ncol;
    const float b2 = ff_b2[n];
    v8f c;
#pragma unroll
    for (int r = 0; r < 8; ++r) c[r] = b2;
#pragma unroll
    for (int kc = 0; kc < 8; ++kc)
      c = wmma16(af[kc], frag_ld(lw2, nt * 8 + kc, lane), c);
#pragma unroll
    for (int r = 0; r < 8; ++r) {
      const int idx = (mbase + r) * Hq + n;
      xs[idx] = xs[idx] + c[r];
    }
  }
  __syncthreads();

  // --- LayerNorm per row (2 lanes/row) ------------------------------------
  {
    const int row  = lane & 15;
    const int half = lane >> 4;
    const float* xr = xs + row * Hq + half * 64;
    float s = 0.f;
#pragma unroll
    for (int j = 0; j < 64; ++j) s += xr[j];
    s += __shfl_xor(s, 16, 32);
    const float mu = s * (1.0f / Hq);
    float v = 0.f;
#pragma unroll
    for (int j = 0; j < 64; ++j) { const float d = xr[j] - mu; v += d * d; }
    v += __shfl_xor(v, 16, 32);
    const float rstd = rsqrtf(v * (1.0f / Hq) + 1e-5f);
    for (int j = 0; j < 64; ++j) {
      const int col = half * 64 + j;
      const int idx = row * Hq + col;
      xs[idx] = (xs[idx] - mu) * rstd * ln_g[col] + ln_b[col];
    }
  }
  __syncthreads();

  // --- k projection -> k_all ----------------------------------------------
  v16h an[4];
#pragma unroll
  for (int kc = 0; kc < 4; ++kc) an[kc] = frag_a_f32(xs, Hq, kc * 32, lane);
  for (int nt = 0; nt < 8; ++nt) {
    v8f c = {};
#pragma unroll
    for (int kc = 0; kc < 4; ++kc)
      c = wmma16(an[kc], frag_ld(lkp, nt * 4 + kc, lane), c);
    const int n = nt * 16 + ncol;
#pragma unroll
    for (int r = 0; r < 8; ++r)
      k_all[(size_t)(tok0 + mbase + r) * Hq + n] = c[r];
  }
}

// ---------------------------------------------------------------------------
// Kernel B: sequential gated-memory scan.  One block per batch, 256 threads:
// 2 threads per M row (64-FMA half-dots combined via shfl_xor(.,1)); k kept
// unnormalized in LDS with 1/||k|| folded into scalars -> 3 barriers/step.
// LDS k reads vectorized as float4 (ds_load_b128) on the serial path.
// ---------------------------------------------------------------------------
__global__ void scan_kernel(const float* __restrict__ k_all,
                            const float* __restrict__ rp_w,
                            const float* __restrict__ rp_b,
                            float* __restrict__ r_out) {
  __shared__ __align__(16) float kv[Hq];
  __shared__ float redA[8];
  __shared__ float redB[8];
  const int b    = blockIdx.x;
  const int tid  = threadIdx.x;   // 0..255
  const int lane = tid & 31;
  const int wid  = tid >> 5;
  const int row  = tid >> 1;      // 0..127
  const int h    = tid & 1;       // half of the row

  float mrow[64];                 // M[row][h*64 .. h*64+63]
#pragma unroll
  for (int j = 0; j < 64; ++j) mrow[j] = 0.f;

  const float* kb = k_all + (size_t)b * Lq * Hq;

  for (int t = 0; t < Lq - 1; ++t) {
    float kq = 0.f;
    if (h == 0) {
      kq = kb[(size_t)t * Hq + row];
      kv[row] = kq;
      __builtin_prefetch(kb + (size_t)(t + 8) * Hq + row, 0, 0);
    }
    float p = kq * kq;            // odd threads contribute 0
#pragma unroll
    for (int off = 16; off > 0; off >>= 1) p += __shfl_xor(p, off, 32);
    if (lane == 0) redA[wid] = p;
    __syncthreads();
    float nk2 = 0.f;
#pragma unroll
    for (int w = 0; w < 8; ++w) nk2 += redA[w];
    const float inv = 1.0f / fmaxf(sqrtf(nk2), 1e-12f);
    // half dot: M[row] . k  (unnormalized), float4 LDS reads
    const float4* k4 = (const float4*)(kv + h * 64);
    float dh = 0.f;
#pragma unroll
    for (int jj = 0; jj < 16; ++jj) {
      const float4 kk = k4[jj];
      dh += mrow[4 * jj + 0] * kk.x + mrow[4 * jj + 1] * kk.y +
            mrow[4 * jj + 2] * kk.z + mrow[4 * jj + 3] * kk.w;
    }
    const float dot = dh + __shfl_xor(dh, 1, 32);
    const float err = kv[row] - dot * inv;
    float e = (h == 0) ? err * err : 0.f;
#pragma unroll
    for (int off = 16; off > 0; off >>= 1) e += __shfl_xor(e, off, 32);
    if (lane == 0) redB[wid] = e;
    __syncthreads();
    float ne2 = 0.f;
#pragma unroll
    for (int w = 0; w < 8; ++w) ne2 += redB[w];
    // fire: ||err|| >= 0.7||k||  <=>  ne2 >= 0.49*nk2
    if (ne2 >= 0.49f * nk2) {
      const float cc = 0.05f * err * inv;   // (1-ALPHA) * err / ||k||
#pragma unroll
      for (int jj = 0; jj < 16; ++jj) {
        const float4 kk = k4[jj];
        mrow[4 * jj + 0] += cc * kk.x;
        mrow[4 * jj + 1] += cc * kk.y;
        mrow[4 * jj + 2] += cc * kk.z;
        mrow[4 * jj + 3] += cc * kk.w;
      }
    }
    __syncthreads();
  }

  // read = M @ q ; r = rp_w @ read + rp_b
  if (h == 0) kv[row] = kb[(size_t)(Lq - 1) * Hq + row];
  __syncthreads();
  {
    const float4* k4 = (const float4*)(kv + h * 64);
    float dh = 0.f;
#pragma unroll
    for (int jj = 0; jj < 16; ++jj) {
      const float4 kk = k4[jj];
      dh += mrow[4 * jj + 0] * kk.x + mrow[4 * jj + 1] * kk.y +
            mrow[4 * jj + 2] * kk.z + mrow[4 * jj + 3] * kk.w;
    }
    const float rd = dh + __shfl_xor(dh, 1, 32);
    __syncthreads();
    if (h == 0) kv[row] = rd;
  }
  __syncthreads();
  {
    const float4* k4 = (const float4*)(kv + h * 64);
    const float4* w4 = (const float4*)(rp_w + (size_t)row * Hq + h * 64);
    float ah = 0.f;
#pragma unroll
    for (int jj = 0; jj < 16; ++jj) {
      const float4 kk = k4[jj];
      const float4 ww = w4[jj];
      ah += ww.x * kk.x + ww.y * kk.y + ww.z * kk.z + ww.w * kk.w;
    }
    const float acc = ah + __shfl_xor(ah, 1, 32);
    if (h == 0) r_out[b * Hq + row] = acc + rp_b[row];
  }
}

// ---------------------------------------------------------------------------
// Kernel C: out = r(32x128) @ out_w^T(32000x128) + out_b, WMMA f16.
// out_w kept f32 (pre-converting would cost more HBM traffic than it saves).
// ---------------------------------------------------------------------------
__global__ void out_gemm_kernel(const float* __restrict__ rmat,
                                const float* __restrict__ out_w,
                                const float* __restrict__ out_b,
                                float* __restrict__ out) {
  const int wave  = threadIdx.x >> 5;
  const int lane  = threadIdx.x & 31;
  const int ntile = blockIdx.x * 4 + wave;
  if (ntile >= Vq / 16) return;
  const int n0    = ntile * 16;
  const int ncol  = lane & 15;
  const int mbase = (lane < 16) ? 0 : 8;

  __builtin_prefetch(out_w + (size_t)(n0 + 64) * Hq, 0, 1);

  v16h a0[4], a1[4];
#pragma unroll
  for (int kc = 0; kc < 4; ++kc) {
    a0[kc] = frag_a_f32(rmat,           Hq, kc * 32, lane);   // rows  0..15
    a1[kc] = frag_a_f32(rmat + 16 * Hq, Hq, kc * 32, lane);   // rows 16..31
  }
  const float bias = out_b[n0 + ncol];
  v8f c0, c1;
#pragma unroll
  for (int r = 0; r < 8; ++r) { c0[r] = bias; c1[r] = bias; }
#pragma unroll
  for (int kc = 0; kc < 4; ++kc) {
    v16h bf = frag_bT_f32(out_w, Hq, n0, kc * 32, lane);
    c0 = wmma16(a0[kc], bf, c0);
    c1 = wmma16(a1[kc], bf, c1);
  }
  const int n = n0 + ncol;
#pragma unroll
  for (int r = 0; r < 8; ++r) {
    out[(size_t)(mbase + r)      * Vq + n] = c0[r];
    out[(size_t)(16 + mbase + r) * Vq + n] = c1[r];
  }
}

// ---------------------------------------------------------------------------
extern "C" void kernel_launch(void* const* d_in, const int* in_sizes, int n_in,
                              void* d_out, int out_size, void* d_ws, size_t ws_size,
                              hipStream_t stream) {
  (void)in_sizes; (void)n_in; (void)out_size; (void)ws_size;
  const int*   seq     = (const int*)d_in[0];
  const float* embed_w = (const float*)d_in[1];
  const float* ff_w1   = (const float*)d_in[2];
  const float* ff_b1   = (const float*)d_in[3];
  const float* ff_w2   = (const float*)d_in[4];
  const float* ff_b2   = (const float*)d_in[5];
  const float* ln_g    = (const float*)d_in[6];
  const float* ln_b    = (const float*)d_in[7];
  const float* kp_w    = (const float*)d_in[8];
  const float* rp_w    = (const float*)d_in[9];
  const float* rp_b    = (const float*)d_in[10];
  const float* out_w   = (const float*)d_in[11];
  const float* out_b   = (const float*)d_in[12];

  // workspace layout
  char* ws = (char*)d_ws;
  float*     k_all = (float*)ws;                                 // 33,554,432 B
  float*     rmat  = (float*)(ws + 33554432);                    //     16,384 B
  _Float16*  w1f   = (_Float16*)(ws + 33570816);                 //     65,536 B
  _Float16*  w2f   = (_Float16*)(ws + 33636352);                 //     65,536 B
  _Float16*  kpf   = (_Float16*)(ws + 33701888);                 //     32,768 B

  // 1) one-time weight swizzle+convert to f16 fragment buffers
  swizzle_w_kernel<<<(256 * 128 + 255) / 256, 256, 0, stream>>>(ff_w1, 256, 128, w1f);
  swizzle_w_kernel<<<(128 * 256 + 255) / 256, 256, 0, stream>>>(ff_w2, 128, 256, w2f);
  swizzle_w_kernel<<<(128 * 128 + 255) / 256, 256, 0, stream>>>(kp_w,  128, 128, kpf);

  // 2) fused token pipeline (TDM-staged weights, WMMA GEMMs)
  fused_token_kernel<<<(Bq * Lq) / (16 * 8), 256, SMEM_A, stream>>>(
      seq, embed_w, w1f, ff_b1, w2f, ff_b2, ln_g, ln_b, kpf, k_all);

  // 3) sequential gated scan
  scan_kernel<<<Bq, 256, 0, stream>>>(k_all, rp_w, rp_b, rmat);

  // 4) vocab projection
  out_gemm_kernel<<<(Vq / 16) / 4, 128, 0, stream>>>(rmat, out_w, out_b,
                                                     (float*)d_out);
}
